// Edge_Drop_Learner_37160057045562
// MI455X (gfx1250) — matrix operand order repaired and checked by reading, not compiled, not run on hardware
//
#include <hip/hip_runtime.h>
#include <math.h>

typedef __attribute__((ext_vector_type(16))) __bf16 bf16x16;
typedef __attribute__((ext_vector_type(8)))  __bf16 bf16x8;
typedef __attribute__((ext_vector_type(8)))  float  f32x8;
typedef __attribute__((ext_vector_type(4)))  float  f32x4;
typedef __attribute__((ext_vector_type(2)))  float  f32x2;

#define GROUP_ROWS 128   // rows (edges/nodes) per workgroup iteration
#define NWAVES 8         // 256 threads
#define HP 18            // padded row pitch (floats) of the H' partial matrix

__device__ __forceinline__ void sched_fence() {
#if __has_builtin(__builtin_amdgcn_sched_barrier)
    __builtin_amdgcn_sched_barrier(0);   // keep load phase and MMA phase separate
#endif
}

// ---------------------------------------------------------------------------
// Cooperative stage: GROUP_ROWS x K fp32 from global -> bf16 in LDS, with a
// 16-byte-chunk XOR swizzle (chunk' = chunk ^ (row & 15)) for bank-conflict-
// free WMMA A-fragment reads. Rows past nrows are clamped (masked later).
// ---------------------------------------------------------------------------
template<int KS>
__device__ __forceinline__ void stage_rows(__bf16* Xs, const float* __restrict__ X,
                                           long long row0, long long nrows) {
    constexpr int K   = 32 * KS;
    constexpr int CPR = K / 8;              // 8-element chunks per row
    constexpr int RPP = 256 / CPR;          // rows staged per pass
    const int t  = threadIdx.x;
    const int c  = t % CPR;
    const int r0 = t / CPR;
#pragma unroll
    for (int p = 0; p < GROUP_ROWS / RPP; ++p) {
        const int row = r0 + p * RPP;
        long long g = row0 + row;
        if (g >= nrows) g = nrows - 1;
        const f32x4* src = (const f32x4*)(X + g * (long long)K + c * 8);
        f32x4 f0 = __builtin_nontemporal_load(src);
        f32x4 f1 = __builtin_nontemporal_load(src + 1);
        bf16x8 b;
#pragma unroll
        for (int i = 0; i < 4; ++i) { b[i] = (__bf16)f0[i]; b[4 + i] = (__bf16)f1[i]; }
        *(bf16x8*)(Xs + row * K + (((c ^ (row & 15)) << 3))) = b;
    }
}

// Prefetch the next group's rows into GL2 (overlaps HBM fetch with WMMAs).
template<int KS>
__device__ __forceinline__ void prefetch_rows(const float* __restrict__ X,
                                              long long row0, long long nrows) {
    constexpr int K = 32 * KS;
    constexpr int LINES = (GROUP_ROWS * K * 4) / (256 * 128);  // 128B lines/thread
    if (row0 + GROUP_ROWS <= nrows) {
        const char* base = (const char*)(X + row0 * (long long)K);
#pragma unroll
        for (int i = 0; i < LINES; ++i)
            __builtin_prefetch(base + ((int)threadIdx.x + i * 256) * 128, 0, 0);
    }
}

// ---------------------------------------------------------------------------
// Per-wave B fragments: wave owns hidden units n = 16*wave + r (lane r).
// Bf[kf] element e = W1[k = 32kf + 8*half + e + (e & 8)][n]  (one-time global reads).
// ---------------------------------------------------------------------------
template<int KS>
__device__ __forceinline__ void load_bfrag(const float* __restrict__ w1, int n, int half,
                                           bf16x16 (&Bf)[KS]) {
#pragma unroll
    for (int kf = 0; kf < KS; ++kf) {
        bf16x16 b;
#pragma unroll
        for (int e = 0; e < 16; ++e)
            b[e] = (__bf16)w1[(32 * kf + 8 * half + e + (e & 8)) * 128 + n];
        Bf[kf] = b;
    }
}

// ---------------------------------------------------------------------------
// A-fragment loads for one 16-row stripe (CDNA5 16-bit A layout: lane (half,r)
// holds row M=r, K runs [32kf+8*half .. +7] and [+16 .. +23]).
// ---------------------------------------------------------------------------
template<int KS>
__device__ __forceinline__ void load_afrags(const __bf16* Xs, int rt, int half, int r,
                                            bf16x16 (&A)[KS]) {
    constexpr int K = 32 * KS;
    const __bf16* rowp = Xs + (16 * rt + r) * K;    // row = 16rt + r ; row&15 == r
#pragma unroll
    for (int kf = 0; kf < KS; ++kf) {
        const int ch0 = 4 * kf + half;              // chunk of run0 ; run1 = ch0+2
        bf16x8 alo = *(const bf16x8*)(rowp + (((ch0)     ^ r) << 3));
        bf16x8 ahi = *(const bf16x8*)(rowp + (((ch0 + 2) ^ r) << 3));
        bf16x16 a;
#pragma unroll
        for (int i = 0; i < 8; ++i) { a[i] = alo[i]; a[8 + i] = ahi[i]; }
        A[kf] = a;
    }
}

template<int KS>
__device__ __forceinline__ f32x8 mma_chain(const bf16x16 (&A)[KS], const bf16x16 (&Bf)[KS]) {
    f32x8 C = {0.f, 0.f, 0.f, 0.f, 0.f, 0.f, 0.f, 0.f};
#pragma unroll
    for (int kf = 0; kf < KS; ++kf)
        C = __builtin_amdgcn_wmma_f32_16x16x32_bf16(false, A[kf], false, Bf[kf],
                                                    (short)0, C, false, false);
    return C;
}

// Epilogue: +b1, LeakyReLU, *w2 slice, then dump per-lane values into the H'
// LDS matrix (pitch HP=18 floats -> conflict-free stores and b64 reads).
// D layout: lane (half,r) holds column n=r, rows M = v + 8*half.
__device__ __forceinline__ void stripe_dump(f32x8 C, float b1v, float w2v,
                                            int half, int r, int rt,
                                            float* __restrict__ hrows) {
#pragma unroll
    for (int v = 0; v < 8; ++v) {
        float h = C[v] + b1v;
        h = (h > 0.f) ? h : 0.01f * h;
        hrows[(16 * rt + v + 8 * half) * HP + r] = h * w2v;
    }
}

// Fixed-order sum of row t of H' across all waves (deterministic).
__device__ __forceinline__ float combine_row(const float* __restrict__ Hp, int t) {
    float s = 0.f;
#pragma unroll
    for (int w = 0; w < NWAVES; ++w) {
        const f32x2* p = (const f32x2*)(Hp + (size_t)(w * GROUP_ROWS + t) * HP);
#pragma unroll
        for (int j = 0; j < 8; ++j) { f32x2 q = p[j]; s += q[0]; s += q[1]; }
    }
    return s;
}

// ---------------------------------------------------------------------------
// Node MLP: out[n] = leaky(X[n,:256] @ W1 + b1) @ w2 + b2
// ---------------------------------------------------------------------------
__global__ __launch_bounds__(256)
void node_mlp_kernel(const float* __restrict__ X, long long Nn,
                     const float* __restrict__ w1, const float* __restrict__ b1,
                     const float* __restrict__ w2, const float* __restrict__ b2,
                     float* __restrict__ out, int numGroups) {
    constexpr int KS = 8;
    extern __shared__ char smem[];
    __bf16* Xs = (__bf16*)smem;                               // 128*256*2 = 64KB
    float*  Hp = (float*)(smem + GROUP_ROWS * 32 * KS * 2);   // 8*128*HP*4 = 72KB

    const int lane = threadIdx.x & 31;
    const int wave = threadIdx.x >> 5;
    const int half = lane >> 4;
    const int r    = lane & 15;
    const int n    = 16 * wave + r;

    bf16x16 Bf[KS];
    load_bfrag<KS>(w1, n, half, Bf);
    const float b1v = b1[n];
    const float w2v = w2[n];
    const float b2v = b2[0];
    float* myh = Hp + wave * GROUP_ROWS * HP;

    for (int g = blockIdx.x; g < numGroups; g += gridDim.x) {
        const long long row0 = (long long)g * GROUP_ROWS;
        stage_rows<KS>(Xs, X, row0, Nn);
        __syncthreads();
        prefetch_rows<KS>(X, row0 + (long long)gridDim.x * GROUP_ROWS, Nn);
#pragma unroll
        for (int rt = 0; rt < GROUP_ROWS / 16; ++rt) {
            bf16x16 A[KS];
            load_afrags<KS>(Xs, rt, half, r, A);
            sched_fence();
            f32x8 C = mma_chain<KS>(A, Bf);
            stripe_dump(C, b1v, w2v, half, r, rt, myh);
        }
        __syncthreads();
        const int t = threadIdx.x;
        if (t < GROUP_ROWS) {
            float s = combine_row(Hp, t) + b2v;
            long long nidx = row0 + t;
            if (nidx < Nn) out[nidx] = s;
        }
        __syncthreads();   // Hp reuse next iteration
    }
}

// ---------------------------------------------------------------------------
// Edge MLP fused with gather + gate + sigmoid + per-group sum of (1 - aug).
// Two stripes per iteration: 16 batched ds_loads, then 8 WMMAs on two
// independent accumulator chains (hides LDS latency + WMMA RAW latency).
// ---------------------------------------------------------------------------
__global__ __launch_bounds__(256)
void edge_gate_kernel(const float* __restrict__ edge_fea, long long E,
                      const float* __restrict__ w1, const float* __restrict__ b1,
                      const float* __restrict__ w2, const float* __restrict__ b2,
                      const int* __restrict__ src, const int* __restrict__ dst,
                      const float* __restrict__ noise,
                      const float* __restrict__ wsv, const float* __restrict__ wdv,
                      float* __restrict__ aug, float* __restrict__ partial, int numGroups) {
    constexpr int KS = 4;
    extern __shared__ char smem[];
    __bf16* Xs   = (__bf16*)smem;                              // 128*128*2 = 32KB
    float*  Hp   = (float*)(smem + GROUP_ROWS * 32 * KS * 2);  // 8*128*HP*4 = 72KB
    float*  wsum = Hp + NWAVES * GROUP_ROWS * HP;              // 8 floats

    const int lane = threadIdx.x & 31;
    const int wave = threadIdx.x >> 5;
    const int half = lane >> 4;
    const int r    = lane & 15;
    const int n    = 16 * wave + r;

    bf16x16 Bf[KS];
    load_bfrag<KS>(w1, n, half, Bf);
    const float b1v = b1[n];
    const float w2v = w2[n];
    const float b2v = b2[0];
    float* myh = Hp + wave * GROUP_ROWS * HP;

    for (int g = blockIdx.x; g < numGroups; g += gridDim.x) {
        const long long row0 = (long long)g * GROUP_ROWS;
        stage_rows<KS>(Xs, edge_fea, row0, E);
        __syncthreads();
        prefetch_rows<KS>(edge_fea, row0 + (long long)gridDim.x * GROUP_ROWS, E);
#pragma unroll
        for (int rt = 0; rt < GROUP_ROWS / 16; rt += 2) {
            bf16x16 A0[KS], A1[KS];
            load_afrags<KS>(Xs, rt,     half, r, A0);
            load_afrags<KS>(Xs, rt + 1, half, r, A1);
            sched_fence();
            f32x8 C0 = mma_chain<KS>(A0, Bf);
            f32x8 C1 = mma_chain<KS>(A1, Bf);
            stripe_dump(C0, b1v, w2v, half, r, rt,     myh);
            stripe_dump(C1, b1v, w2v, half, r, rt + 1, myh);
        }
        __syncthreads();

        // Combine: thread t (< 128) owns edge row0 + t.
        const int t = threadIdx.x;
        float local = 0.f;
        if (t < GROUP_ROWS) {
            float w_e = combine_row(Hp, t) + b2v;
            long long e = row0 + t;
            if (e < E) {
                float weight = w_e + wsv[src[e]] + wdv[dst[e]];
                float u = noise[e];
                // eps = (2*BIAS - 1)*u + (1 - BIAS), BIAS = 1e-4
                float eps  = (2.0f * 1e-4f - 1.0f) * u + (1.0f - 1e-4f);
                float gate = (logf(eps) - log1pf(-eps) + weight) * 2.0f;  // / TEMP(0.5)
                float aw   = 1.0f / (1.0f + expf(-gate));
                __builtin_nontemporal_store(aw, &aug[e]);
                local = 1.0f - aw;
            }
        }
        // Deterministic per-group partial: per-wave shuffle sum -> 8 LDS slots
        // -> thread 0 sums in fixed order.
#pragma unroll
        for (int off = 16; off; off >>= 1) local += __shfl_xor(local, off, 32);
        if (lane == 0) wsum[wave] = local;
        __syncthreads();
        if (threadIdx.x == 0) {
            float s = 0.f;
#pragma unroll
            for (int w = 0; w < NWAVES; ++w) s += wsum[w];
            partial[g] = s;
        }
        __syncthreads();   // wsum/Hp reuse next iteration
    }
}

__global__ __launch_bounds__(256)
void reduce_kernel(const float* __restrict__ partial, int T, float* __restrict__ out0,
                   float invE) {
    __shared__ float sm[256];
    float s = 0.f;
    for (int i = threadIdx.x; i < T; i += 256) s += partial[i];  // fixed order
    sm[threadIdx.x] = s;
    __syncthreads();
    for (int st = 128; st; st >>= 1) {
        if ((int)threadIdx.x < st) sm[threadIdx.x] += sm[threadIdx.x + st];
        __syncthreads();
    }
    if (threadIdx.x == 0) out0[0] = sm[0] * invE;
}

extern "C" void kernel_launch(void* const* d_in, const int* in_sizes, int n_in,
                              void* d_out, int out_size, void* d_ws, size_t ws_size,
                              hipStream_t stream) {
    const float* node_emb = (const float*)d_in[0];
    const float* edge_fea = (const float*)d_in[1];
    const float* noise    = (const float*)d_in[2];
    const int*   src      = (const int*)d_in[3];
    const int*   dst      = (const int*)d_in[4];
    const float* w_src1   = (const float*)d_in[5];
    const float* b_src1   = (const float*)d_in[6];
    const float* w_src2   = (const float*)d_in[7];
    const float* b_src2   = (const float*)d_in[8];
    const float* w_dst1   = (const float*)d_in[9];
    const float* b_dst1   = (const float*)d_in[10];
    const float* w_dst2   = (const float*)d_in[11];
    const float* b_dst2   = (const float*)d_in[12];
    const float* w_edge1  = (const float*)d_in[13];
    const float* b_edge1  = (const float*)d_in[14];
    const float* w_edge2  = (const float*)d_in[15];
    const float* b_edge2  = (const float*)d_in[16];

    const long long Nn = (long long)in_sizes[0] / 256;   // D = 256
    const long long E  = (long long)in_sizes[2];         // noise is [E]
    const int groupsN = (int)((Nn + GROUP_ROWS - 1) / GROUP_ROWS);
    const int groupsE = (int)((E + GROUP_ROWS - 1) / GROUP_ROWS);

    // Workspace: [partials groupsE][w_s N][w_d N], 128-float aligned.
    float* partial = (float*)d_ws;
    float* wsv = partial + (((size_t)groupsE + 127) & ~(size_t)127);
    float* wdv = wsv + (((size_t)Nn + 127) & ~(size_t)127);

    const size_t hpBytes = (size_t)NWAVES * GROUP_ROWS * HP * 4 + 64;
    const size_t ldsNode = (size_t)GROUP_ROWS * 256 * 2 + hpBytes;   // ~139KB
    const size_t ldsEdge = (size_t)GROUP_ROWS * 128 * 2 + hpBytes;   // ~106KB

    int blocksN = groupsN;                 // 391 for N = 50000
    int blocksE = groupsE < 2048 ? groupsE : 2048;

    float* out = (float*)d_out;   // out[0] = reg, out[1..E] = aug_edge_weight

    node_mlp_kernel<<<blocksN, 256, ldsNode, stream>>>(node_emb, Nn, w_src1, b_src1,
                                                       w_src2, b_src2, wsv, groupsN);
    node_mlp_kernel<<<blocksN, 256, ldsNode, stream>>>(node_emb, Nn, w_dst1, b_dst1,
                                                       w_dst2, b_dst2, wdv, groupsN);
    edge_gate_kernel<<<blocksE, 256, ldsEdge, stream>>>(edge_fea, E, w_edge1, b_edge1,
                                                        w_edge2, b_edge2, src, dst, noise,
                                                        wsv, wdv, out + 1, partial, groupsE);
    reduce_kernel<<<1, 256, 0, stream>>>(partial, groupsE, out, 1.0f / (float)E);
}
